// SimilarityLoss_28209345200703
// MI455X (gfx1250) — matrix operand order repaired and checked by reading, not compiled, not run on hardware
//
#include <hip/hip_runtime.h>
#include <hip/hip_bf16.h>

typedef _Float16 v16h __attribute__((ext_vector_type(16)));
typedef _Float16 h8   __attribute__((ext_vector_type(8)));
typedef float    v8f  __attribute__((ext_vector_type(8)));

#define N_ROWS 8192
#define DIM    256
#define TILE   128
#define KC     32
#define LDS_ST 40   // f16 stride per staged row: 80B (16B-aligned, bank-friendly)
#define PANEL  (TILE * LDS_ST)

// ---- CDNA5 async-to-LDS path (GLOBAL_LOAD_ASYNC_TO_LDS_B128 / ASYNCcnt) ----
#if defined(__has_builtin)
#  if __has_builtin(__builtin_amdgcn_global_load_async_to_lds_b128)
#    define HAVE_ASYNC_LDS 1
#  endif
#  if __has_builtin(__builtin_amdgcn_s_wait_asynccnt)
#    define HAVE_ASYNC_WAIT 1
#  endif
#endif
#ifndef HAVE_ASYNC_LDS
#  define HAVE_ASYNC_LDS 0
#endif
#ifndef HAVE_ASYNC_WAIT
#  define HAVE_ASYNC_WAIT 0
#endif

#if HAVE_ASYNC_LDS
// Param 0 of the builtin is 'int __vector_size__(16) * ' in AS(1) (global);
// destination is the matching 128-bit vector pointer in AS(3) (LDS).
typedef int i4v __attribute__((vector_size(16)));
typedef __attribute__((address_space(1))) i4v g_i4v;
typedef __attribute__((address_space(3))) i4v l_i4v;
#  if HAVE_ASYNC_WAIT
#    define WAIT_ASYNC(n) __builtin_amdgcn_s_wait_asynccnt(n)
#  else
#    define WAIT_ASYNC(n) asm volatile("s_wait_asynccnt " #n ::: "memory")
#  endif
#else
#  define WAIT_ASYNC(n) ((void)0)
#endif

// Copy one 16B chunk global -> LDS.
__device__ __forceinline__ void copy16(_Float16* ldst, const _Float16* gsrc) {
#if HAVE_ASYNC_LDS
    __builtin_amdgcn_global_load_async_to_lds_b128(
        (g_i4v*)gsrc, (l_i4v*)ldst, /*offset=*/0, /*cpol=*/0);
#else
    *(h8*)ldst = *(const h8*)gsrc;
#endif
}

// ---------------------------------------------------------------------------
// Pass 1: L2-normalize rows (clamp at eps=1e-8 like torch) and cast to f16.
// One wave per row: 32 lanes x 8 floats = 256 = D.
// ---------------------------------------------------------------------------
__global__ void __launch_bounds__(256)
norm_cast_kernel(const float* __restrict__ o, const float* __restrict__ t,
                 _Float16* __restrict__ o16, _Float16* __restrict__ t16) {
    int wave = threadIdx.x >> 5;
    int lane = threadIdx.x & 31;
    int row_lin = blockIdx.x * 8 + wave;

    const float* src;
    _Float16* dst;
    int row;
    if (row_lin < N_ROWS) { src = o; dst = o16; row = row_lin; }
    else                  { src = t; dst = t16; row = row_lin - N_ROWS; }

    const float4* p = (const float4*)(src + (size_t)row * DIM + lane * 8);
    float4 x0 = p[0];
    float4 x1 = p[1];

    float ss = x0.x*x0.x + x0.y*x0.y + x0.z*x0.z + x0.w*x0.w
             + x1.x*x1.x + x1.y*x1.y + x1.z*x1.z + x1.w*x1.w;
    #pragma unroll
    for (int off = 16; off >= 1; off >>= 1)
        ss += __shfl_xor(ss, off, 32);

    float scale = 1.0f / fmaxf(sqrtf(ss), 1e-8f);

    h8 hv;
    hv[0] = (_Float16)(x0.x * scale);
    hv[1] = (_Float16)(x0.y * scale);
    hv[2] = (_Float16)(x0.z * scale);
    hv[3] = (_Float16)(x0.w * scale);
    hv[4] = (_Float16)(x1.x * scale);
    hv[5] = (_Float16)(x1.y * scale);
    hv[6] = (_Float16)(x1.z * scale);
    hv[7] = (_Float16)(x1.w * scale);
    *(h8*)(dst + (size_t)row * DIM + lane * 8) = hv;
}

// ---------------------------------------------------------------------------
// Stage 4 panels (o/t x A/B side) of 128 rows x 32 f16 into one LDS buffer.
// 512 x 16B chunks per panel, 256 threads -> 2 chunks/thread/panel,
// 8 (async) copies per thread total.
// ---------------------------------------------------------------------------
__device__ __forceinline__ void stage_panels(
    _Float16* buf, const _Float16* __restrict__ o16,
    const _Float16* __restrict__ t16, size_t baseA, size_t baseB,
    int kc, int tid) {
    _Float16* oA = buf;
    _Float16* oB = buf + 1 * PANEL;
    _Float16* tA = buf + 2 * PANEL;
    _Float16* tB = buf + 3 * PANEL;
    #pragma unroll
    for (int c = tid; c < 512; c += 256) {
        int r = c >> 2;          // staged row 0..127
        int q = c & 3;           // 16B chunk within row
        size_t goff = (size_t)r * DIM + (size_t)kc * KC + q * 8;
        int    loff = r * LDS_ST + q * 8;
        copy16(oA + loff, o16 + baseA + goff);
        copy16(oB + loff, o16 + baseB + goff);
        copy16(tA + loff, t16 + baseA + goff);
        copy16(tB + loff, t16 + baseB + goff);
    }
}

// ---------------------------------------------------------------------------
// Pass 2: fused dual-GEMM + masked squared-difference reduction.
// Upper-triangular 128x128 tiles only; off-diagonal tiles weighted 2x.
// 8 waves (4x2) per block; each wave computes a 32x64 patch for BOTH matrices
// via v_wmma_f32_16x16x32_f16. Double-buffered async-to-LDS staging.
// ---------------------------------------------------------------------------
__global__ void __launch_bounds__(256)
gemm_loss_kernel(const _Float16* __restrict__ o16,
                 const _Float16* __restrict__ t16,
                 double* __restrict__ acc) {
    int jb = blockIdx.x;   // column tile
    int ib = blockIdx.y;   // row tile
    if (ib > jb) return;   // block-uniform: symmetry, compute upper triangle

    __shared__ _Float16 smem[2 * 4 * PANEL];   // double-buffered, 80 KB
    __shared__ float wsum[8];

    const int tid  = threadIdx.x;
    const int lane = tid & 31;
    const int wave = tid >> 5;
    const int half = lane >> 4;    // wave32 half-select per WMMA layouts
    const int l16  = lane & 15;
    const int wm   = wave >> 1;    // 0..3 -> 32-row strip
    const int wn   = wave & 1;     // 0..1 -> 64-col strip

    v8f acc_o[2][4];
    v8f acc_t[2][4];
    #pragma unroll
    for (int mi = 0; mi < 2; ++mi)
        #pragma unroll
        for (int ni = 0; ni < 4; ++ni) {
            acc_o[mi][ni] = (v8f){0.f,0.f,0.f,0.f,0.f,0.f,0.f,0.f};
            acc_t[mi][ni] = (v8f){0.f,0.f,0.f,0.f,0.f,0.f,0.f,0.f};
        }

    const size_t baseA = (size_t)ib * TILE * DIM;
    const size_t baseB = (size_t)jb * TILE * DIM;

    // Prologue: stage chunk 0 into buffer 0.
    stage_panels(smem, o16, t16, baseA, baseB, 0, tid);

    for (int kc = 0; kc < DIM / KC; ++kc) {
        _Float16* cur = smem + (kc & 1) * (4 * PANEL);
        _Float16* nxt = smem + ((kc + 1) & 1) * (4 * PANEL);

        // Kick off next chunk's staging; overlaps with this chunk's WMMAs.
        if (kc + 1 < DIM / KC) {
            stage_panels(nxt, o16, t16, baseA, baseB, kc + 1, tid);
            WAIT_ASYNC(8);   // the 8 just-issued may fly; current chunk done
        } else {
            WAIT_ASYNC(0);
        }
        __syncthreads();     // current buffer visible to all waves

        const _Float16* oA = cur;
        const _Float16* oB = cur + 1 * PANEL;
        const _Float16* tA = cur + 2 * PANEL;
        const _Float16* tB = cur + 3 * PANEL;

        // Two passes (o then t) to keep live fragment VGPRs low.
        #pragma unroll
        for (int pass = 0; pass < 2; ++pass) {
            const _Float16* A = pass ? tA : oA;
            const _Float16* B = pass ? tB : oB;

            v16h af[2];
            v16h bf[4];
            // A fragment (16x32): lane half selects k-phase; two 16B reads.
            #pragma unroll
            for (int mi = 0; mi < 2; ++mi) {
                int r = wm * 32 + mi * 16 + l16;
                const _Float16* p = A + r * LDS_ST + half * 8;
                h8 lo = *(const h8*)(p);       // k = 8*half + 0..7
                h8 hi = *(const h8*)(p + 16);  // k = 16 + 8*half + 0..7
                af[mi] = __builtin_shufflevector(lo, hi,
                    0,1,2,3,4,5,6,7,8,9,10,11,12,13,14,15);
            }
            // B fragment (32x16): column = lane&15; two 16B reads, same k order as A.
            #pragma unroll
            for (int ni = 0; ni < 4; ++ni) {
                int r = wn * 64 + ni * 16 + l16;
                const _Float16* p = B + r * LDS_ST + half * 8;
                h8 lo = *(const h8*)(p);       // k = 8*half + 0..7
                h8 hi = *(const h8*)(p + 16);  // k = 16 + 8*half + 0..7
                bf[ni] = __builtin_shufflevector(lo, hi,
                    0,1,2,3,4,5,6,7,8,9,10,11,12,13,14,15);
            }
            #pragma unroll
            for (int mi = 0; mi < 2; ++mi)
                #pragma unroll
                for (int ni = 0; ni < 4; ++ni) {
                    if (pass == 0)
                        acc_o[mi][ni] = __builtin_amdgcn_wmma_f32_16x16x32_f16(
                            false, af[mi], false, bf[ni],
                            (short)0, acc_o[mi][ni], false, false);
                    else
                        acc_t[mi][ni] = __builtin_amdgcn_wmma_f32_16x16x32_f16(
                            false, af[mi], false, bf[ni],
                            (short)0, acc_t[mi][ni], false, false);
                }
        }
        __syncthreads();     // all reads of `cur` done before it is re-staged
    }

    // Fused epilogue: (sim_o - sim_t)^2, diagonal masked, per-lane partial.
    float lsum = 0.0f;
    #pragma unroll
    for (int mi = 0; mi < 2; ++mi)
        #pragma unroll
        for (int ni = 0; ni < 4; ++ni)
            #pragma unroll
            for (int e = 0; e < 8; ++e) {
                int rg = ib * TILE + wm * 32 + mi * 16 + e + half * 8;
                int cg = jb * TILE + wn * 64 + ni * 16 + l16;
                float d = acc_o[mi][ni][e] - acc_t[mi][ni][e];
                lsum += (rg == cg) ? 0.0f : d * d;
            }
    #pragma unroll
    for (int off = 16; off >= 1; off >>= 1)
        lsum += __shfl_xor(lsum, off, 32);
    if (lane == 0) wsum[wave] = lsum;
    __syncthreads();
    if (wave == 0) {
        float s = (lane < 8) ? wsum[lane] : 0.0f;
        #pragma unroll
        for (int off = 16; off >= 1; off >>= 1)
            s += __shfl_xor(s, off, 32);
        if (lane == 0) {
            float w = (ib == jb) ? 1.0f : 2.0f;   // symmetry doubling
            atomicAdd(acc, (double)(s * w));
        }
    }
}

// ---------------------------------------------------------------------------
__global__ void zero_acc_kernel(double* acc) { acc[0] = 0.0; }

__global__ void finalize_kernel(const double* __restrict__ acc,
                                const float* __restrict__ weight,
                                float* __restrict__ out) {
    out[0] = (float)(acc[0] * (double)weight[0] /
                     ((double)N_ROWS * (double)N_ROWS));
}

// ---------------------------------------------------------------------------
extern "C" void kernel_launch(void* const* d_in, const int* in_sizes, int n_in,
                              void* d_out, int out_size, void* d_ws, size_t ws_size,
                              hipStream_t stream) {
    const float* o = (const float*)d_in[0];
    const float* t = (const float*)d_in[1];
    const float* w = (const float*)d_in[2];
    float* out = (float*)d_out;

    double*   acc = (double*)d_ws;
    _Float16* o16 = (_Float16*)((char*)d_ws + 16);
    _Float16* t16 = o16 + (size_t)N_ROWS * DIM;

    zero_acc_kernel<<<1, 1, 0, stream>>>(acc);
    norm_cast_kernel<<<2 * N_ROWS / 8, 256, 0, stream>>>(o, t, o16, t16);
    dim3 grid(N_ROWS / TILE, N_ROWS / TILE);   // 64 x 64, lower triangle exits
    gemm_loss_kernel<<<grid, 256, 0, stream>>>(o16, t16, acc);
    finalize_kernel<<<1, 1, 0, stream>>>(acc, w, out);
}
